// EntmaxAttention_16355235463944
// MI455X (gfx1250) — compile-verified, matrix operand order checked by reading, not run on hardware
//
#include <hip/hip_runtime.h>
#include <math.h>

// ---------------------------------------------------------------------------
// Problem constants (reference: B=64, S=2048, H=512, A=128)
// ---------------------------------------------------------------------------
#define BB 64
#define SS 2048
#define HH 512
#define AA 128

typedef __attribute__((ext_vector_type(16))) __bf16 v16bf;
typedef __attribute__((ext_vector_type(8)))  float  v8f;

static __device__ __forceinline__ __bf16 f2bf(float f) {
    return (__bf16)f;
}

// ---------------------------------------------------------------------------
// Kernel 1: scores[b,s] = tanh(lstm[b,s,:] @ W1 + b1) @ W2 + b2
// GEMM [131072 x 512] x [512 x 128] on v_wmma_f32_16x16x32_bf16.
// Block = 256 threads = 8 waves; each wave computes 16 rows x 128 cols.
// W1 staged transposed (Wt[n][k]) in LDS as bf16, K-chunked by 128.
// ---------------------------------------------------------------------------
#define KCHUNK 128
#define WT_PAD 144   // 144 halves = 288B row stride -> 32B-aligned v16bf loads

__global__ void __launch_bounds__(256)
entmax_scores_kernel(const float* __restrict__ lstm,
                     const float* __restrict__ W1,
                     const float* __restrict__ b1,
                     const float* __restrict__ W2,
                     const float* __restrict__ b2,
                     float* __restrict__ scores) {
    __shared__ __align__(64) __bf16 Wt[AA][WT_PAD];

    const int tid      = threadIdx.x;
    const int lane     = tid & 31;
    const int waveId   = tid >> 5;
    const int l16      = lane & 15;
    const int half     = lane >> 4;                 // 0: K 0..15, 1: K 16..31
    const int rowBase  = blockIdx.x * 128 + waveId * 16;
    const int aRow     = rowBase + l16;             // A-fragment row for this lane

    v8f acc[8];
    const v8f vzero = {0.f, 0.f, 0.f, 0.f, 0.f, 0.f, 0.f, 0.f};
#pragma unroll
    for (int t = 0; t < 8; ++t) acc[t] = vzero;

    const float* aPtrRow = lstm + (size_t)aRow * HH;

    for (int kc = 0; kc < HH / KCHUNK; ++kc) {
        // --- stage W1 chunk transposed into LDS (bf16) ---
        __syncthreads();
#pragma unroll
        for (int i = 0; i < (KCHUNK * AA) / 256; ++i) {
            int idx = tid + i * 256;
            int k   = idx >> 7;        // 0..127 within chunk
            int n   = idx & 127;       // output column
            Wt[n][k] = f2bf(W1[(size_t)(kc * KCHUNK + k) * AA + n]);
        }
        __syncthreads();

        // --- 4 K-steps of 32 within the chunk ---
#pragma unroll
        for (int ks = 0; ks < KCHUNK / 32; ++ks) {
            const int kloc  = ks * 32 + half * 16;          // lane's 16-K slice
            const int kglob = kc * KCHUNK + kloc;

            // A fragment: 16 consecutive K values of this lane's row
            float buf[16];
            const float4* ap4 = reinterpret_cast<const float4*>(aPtrRow + kglob);
            *reinterpret_cast<float4*>(&buf[0])  = ap4[0];
            *reinterpret_cast<float4*>(&buf[4])  = ap4[1];
            *reinterpret_cast<float4*>(&buf[8])  = ap4[2];
            *reinterpret_cast<float4*>(&buf[12]) = ap4[3];
            v16bf af;
#pragma unroll
            for (int i = 0; i < 16; ++i) af[i] = f2bf(buf[i]);

            // 8 N-tiles of 16 columns each
#pragma unroll
            for (int t = 0; t < 8; ++t) {
                const v16bf bf =
                    *reinterpret_cast<const v16bf*>(&Wt[16 * t + l16][kloc]);
                acc[t] = __builtin_amdgcn_wmma_f32_16x16x32_bf16(
                    /*neg_a=*/false, af, /*neg_b=*/false, bf,
                    /*c_mod=*/(short)0, acc[t],
                    /*reuse_a=*/false, /*reuse_b=*/false);
            }
        }
    }

    // --- epilogue: tanh, dot with W2, add biases, cross-lane reduce ---
    float b1c[8], w2c[8];
#pragma unroll
    for (int t = 0; t < 8; ++t) {
        b1c[t] = b1[16 * t + l16];
        w2c[t] = W2[16 * t + l16];
    }
    const float b2v = b2[0];

#pragma unroll
    for (int r = 0; r < 8; ++r) {
        float sum = 0.f;
#pragma unroll
        for (int t = 0; t < 8; ++t)
            sum += tanhf(acc[t][r] + b1c[t]) * w2c[t];
        // reduce across the 16-lane half (rows r and r+8 live in the two halves)
#pragma unroll
        for (int off = 1; off < 16; off <<= 1)
            sum += __shfl_xor(sum, off, 32);
        if (l16 == 0)
            scores[rowBase + r + 8 * half] = sum + b2v;
    }
}

// ---------------------------------------------------------------------------
// Kernel 2: exact entmax-1.5 over S=2048 per batch row. One block per b.
// Bitonic sort (descending) + dual inclusive scans in LDS.
// ---------------------------------------------------------------------------
#define ETH 256                 // threads
#define ITEMS (SS / ETH)        // 8 per thread

__global__ void __launch_bounds__(ETH)
entmax15_kernel(const float* __restrict__ scores,
                float* __restrict__ weights) {
    __shared__ float sS[SS];    // shifted scores (original order)
    __shared__ float srt[SS];   // sorted descending
    __shared__ float cs1[SS];   // cumsum
    __shared__ float cs2[SS];   // cumsum of squares
    __shared__ float red[ETH];
    __shared__ float redB[ETH];
    __shared__ float tauStarSh;
    __shared__ int   supportSh;

    const int b   = blockIdx.x;
    const int tid = threadIdx.x;
    const float* src = scores + (size_t)b * SS;

    // load + max reduce
    float m = -3.4e38f;
    for (int i = tid; i < SS; i += ETH) {
        float v = src[i];
        sS[i] = v;
        m = fmaxf(m, v);
    }
    red[tid] = m;
    __syncthreads();
    for (int s = ETH / 2; s > 0; s >>= 1) {
        if (tid < s) red[tid] = fmaxf(red[tid], red[tid + s]);
        __syncthreads();
    }
    const float smax = red[0];
    __syncthreads();

    // shift by max, divide by 2, copy into sort buffer
    for (int i = tid; i < SS; i += ETH) {
        float v = (sS[i] - smax) * 0.5f;
        sS[i]  = v;
        srt[i] = v;
    }
    if (tid == 0) supportSh = 0;
    __syncthreads();

    // bitonic sort, descending
    for (int k = 2; k <= SS; k <<= 1) {
        for (int j = k >> 1; j > 0; j >>= 1) {
            for (int i = tid; i < SS; i += ETH) {
                int ixj = i ^ j;
                if (ixj > i) {
                    float a = srt[i], c = srt[ixj];
                    bool up = ((i & k) == 0);
                    // descending overall: invert ascending swap condition
                    if (up ? (a < c) : (a > c)) {
                        srt[i] = c;
                        srt[ixj] = a;
                    }
                }
            }
            __syncthreads();
        }
    }

    // dual inclusive scan: per-thread serial over ITEMS, then block offsets
    const int base = tid * ITEMS;
    float r1 = 0.f, r2 = 0.f;
#pragma unroll
    for (int j = 0; j < ITEMS; ++j) {
        float v = srt[base + j];
        r1 += v;
        r2 += v * v;
        cs1[base + j] = r1;
        cs2[base + j] = r2;
    }
    red[tid]  = r1;
    redB[tid] = r2;
    __syncthreads();
    if (tid == 0) {
        float o1 = 0.f, o2 = 0.f;
        for (int t = 0; t < ETH; ++t) {
            float t1 = red[t],  t2 = redB[t];
            red[t]  = o1;  redB[t] = o2;
            o1 += t1;      o2 += t2;
        }
    }
    __syncthreads();
    const float o1 = red[tid], o2 = redB[tid];
#pragma unroll
    for (int j = 0; j < ITEMS; ++j) {
        cs1[base + j] += o1;
        cs2[base + j] += o2;
    }
    __syncthreads();

    // support size: count(tau_i <= srt_i)
    int cnt = 0;
#pragma unroll
    for (int j = 0; j < ITEMS; ++j) {
        int i = base + j;
        float rho   = (float)(i + 1);
        float mean  = cs1[i] / rho;
        float msq   = cs2[i] / rho;
        float ss    = rho * (msq - mean * mean);
        float delta = (1.0f - ss) / rho;
        float tau   = mean - sqrtf(fmaxf(delta, 0.0f));
        cnt += (tau <= srt[i]) ? 1 : 0;
    }
    atomicAdd(&supportSh, cnt);
    __syncthreads();
    if (tid == 0) {
        int i = supportSh - 1;
        float rho   = (float)(i + 1);
        float mean  = cs1[i] / rho;
        float msq   = cs2[i] / rho;
        float ss    = rho * (msq - mean * mean);
        float delta = (1.0f - ss) / rho;
        tauStarSh   = mean - sqrtf(fmaxf(delta, 0.0f));
    }
    __syncthreads();
    const float tauStar = tauStarSh;

    // weights = clip(s - tau*, 0)^2 in original order
    float* dst = weights + (size_t)b * SS;
    for (int i = tid; i < SS; i += ETH) {
        float w = fmaxf(sS[i] - tauStar, 0.0f);
        dst[i] = w * w;
    }
}

// ---------------------------------------------------------------------------
// Kernel 3: context[b,h] = sum_s lstm[b,s,h] * weights[b,s]
// One block per batch, 512 threads (one per h), weights staged via LDS.
// ---------------------------------------------------------------------------
__global__ void __launch_bounds__(512)
entmax_context_kernel(const float* __restrict__ lstm,
                      const float* __restrict__ weights,
                      float* __restrict__ context) {
    __shared__ float wsh[512];
    const int b   = blockIdx.x;
    const int tid = threadIdx.x;

    const float* lb = lstm + (size_t)b * SS * HH;
    const float* wb = weights + (size_t)b * SS;

    float acc = 0.f;
    for (int s0 = 0; s0 < SS; s0 += 512) {
        __syncthreads();
        wsh[tid] = wb[s0 + tid];
        __syncthreads();
#pragma unroll 4
        for (int j = 0; j < 512; ++j)
            acc = fmaf(lb[(size_t)(s0 + j) * HH + tid], wsh[j], acc);
    }
    context[(size_t)b * HH + tid] = acc;
}

// ---------------------------------------------------------------------------
// Launch
// ---------------------------------------------------------------------------
extern "C" void kernel_launch(void* const* d_in, const int* in_sizes, int n_in,
                              void* d_out, int out_size, void* d_ws, size_t ws_size,
                              hipStream_t stream) {
    const float* lstm = (const float*)d_in[0];   // [B,S,H]
    const float* W1   = (const float*)d_in[1];   // [H,A]
    const float* b1   = (const float*)d_in[2];   // [A]
    const float* W2   = (const float*)d_in[3];   // [A,1]
    const float* b2   = (const float*)d_in[4];   // [1]

    float* out      = (float*)d_out;
    float* context  = out;                       // [B,H]  = 32768 floats
    float* weights  = out + (size_t)BB * HH;     // [B,S]  = 131072 floats
    float* scores   = (float*)d_ws;              // [B,S] scratch

    (void)in_sizes; (void)n_in; (void)out_size; (void)ws_size;

    // 131072 rows / 128 rows per block = 1024 blocks
    entmax_scores_kernel<<<(BB * SS) / 128, 256, 0, stream>>>(
        lstm, W1, b1, W2, b2, scores);

    entmax15_kernel<<<BB, ETH, 0, stream>>>(scores, weights);

    entmax_context_kernel<<<BB, 512, 0, stream>>>(lstm, weights, context);
}
